// LocalAttention_67860483276981
// MI455X (gfx1250) — matrix operand order repaired and checked by reading, compiled, not run on hardware
//
#include <hip/hip_runtime.h>
#include <stdint.h>

// Fused windowed attention for MI455X (gfx1250, wave32, WMMA bf16).
// One workgroup (256 threads = 8 waves) per 7x7 window; all intermediates in LDS.
// Each wave owns a fixed M-tile per stage -> A-fragments preloaded once into
// registers, inner loops are pure {B-load clause + WMMA chain}.

#define WSZ   7
#define Bc    16
#define Cc    256
#define Hc    112
#define Wc    112
#define TOK   49
#define TP    64          // tokens padded to 4 M-tiles
#define SX    264         // token-major bf16 row stride (elems): 132 dw == 4 mod 64 banks
#define SV    72          // V^T row stride (tokens, elems)
#define SS    68          // score row stride (f32)
#define SP    64          // P row stride (bf16 elems)

#define WQKV_OFF   0
#define WPROJ_OFF  196608
#define BQKV_OFF   262144
#define BPROJ_OFF  262912
#define WTS_ELEMS  263168

typedef __attribute__((ext_vector_type(16))) __bf16 v16bf;
typedef __attribute__((ext_vector_type(8)))  float  v8f;

union FragU { uint4 q[2]; v16bf v; };

__device__ __forceinline__ uint16_t f2bf(float f) {
  union { float f; uint32_t u; } c; c.f = f;
  uint32_t u = c.u;
  return (uint16_t)((u + 0x7FFFu + ((u >> 16) & 1u)) >> 16);   // RNE
}
__device__ __forceinline__ float bf2f(uint16_t h) {
  union { uint32_t u; float f; } c; c.u = ((uint32_t)h) << 16;
  return c.f;
}

// A-matrix 16x32 bf16 frag: lanes 0-15 -> M=lane, K {0..7,16..23}; lanes 16-31 -> K {8..15,24..31}
__device__ __forceinline__ v16bf loadA(const uint16_t* mat, int stride, int mBase, int kBase, int lane) {
  const uint16_t* r = mat + (mBase + (lane & 15)) * stride + kBase + ((lane >> 4) << 3);
  FragU f;
  f.q[0] = *(const uint4*)r;
  f.q[1] = *(const uint4*)(r + 16);
  return f.v;
}
// B-matrix 32x16 bf16 frag: lane -> column N=lane&15; lanes 0-15 K {0..7,16..23}; lanes 16-31 K {8..15,24..31}
// Source rows must be contiguous along K (row index = N).
__device__ __forceinline__ v16bf loadB(const uint16_t* mat, int stride, int nBase, int kBase, int lane) {
  const uint16_t* r = mat + (nBase + (lane & 15)) * stride + kBase + ((lane >> 4) << 3);
  FragU f;
  f.q[0] = *(const uint4*)r;
  f.q[1] = *(const uint4*)(r + 16);
  return f.v;
}

__global__ __launch_bounds__(256)
void la_prep_weights(const float* __restrict__ wqkv, const float* __restrict__ bqkv,
                     const float* __restrict__ wproj, const float* __restrict__ bproj,
                     uint16_t* __restrict__ wts) {
  int i = blockIdx.x * blockDim.x + threadIdx.x;
  if (i >= WTS_ELEMS) return;
  float v;
  if (i < WPROJ_OFF)          v = wqkv[i];
  else if (i < BQKV_OFF)      v = wproj[i - WPROJ_OFF];
  else if (i < BPROJ_OFF)     v = bqkv[i - BQKV_OFF];
  else                        v = bproj[i - BPROJ_OFF];
  wts[i] = f2bf(v);
}

__global__ __launch_bounds__(256)
void la_window_attn(const float* __restrict__ x,
                    const uint16_t* __restrict__ wts,
                    float* __restrict__ out) {
  extern __shared__ char smem[];
  uint16_t* xw = (uint16_t*)smem;                 // [64][264] bf16, reused as O
  uint16_t* Qm = xw + TP * SX;                    // [64][264]
  uint16_t* Km = Qm + TP * SX;                    // [64][264]
  uint16_t* Vt = Km + TP * SX;                    // [256][72]  (V transposed: channel-major)
  float*    Sm = (float*)(Vt + Cc * SV);          // [64][68] f32 scores
  uint16_t* Pm = (uint16_t*)(Sm + TP * SS);       // [64][64] bf16 softmax probs
  // total = 3*33792 + 36864 + 17408 + 8192 = 163840 B -> 2 WGs per 320KB WGP

  const int tid  = threadIdx.x;
  const int lane = tid & 31;
  const int wv   = tid >> 5;
  const int mtF  = wv & 3;                        // wave-fixed M-tile (rows mtF*16..+15)
  const int ntP  = wv >> 2;                       // N-tile parity handled by this wave

  const int wid = blockIdx.x;                     // 4096 windows
  const int b   = wid >> 8;
  const int wh  = (wid >> 4) & 15;
  const int ww  = wid & 15;
  const int h0  = wh * WSZ;
  const int w0  = ww * WSZ;

  // ---- Stage 0: gather window of x -> LDS bf16 (pad rows zeroed) ----
  for (int idx = tid; idx < TP * Cc; idx += 256) {
    int m = idx & 63;                             // consecutive lanes -> consecutive pixels
    int c = idx >> 6;
    float v = 0.0f;
    if (m < TOK) {
      int hh = h0 + m / WSZ;
      int w2 = w0 + m % WSZ;
      v = x[((b * Cc + c) * Hc + hh) * Wc + w2];
    }
    xw[m * SX + c] = f2bf(v);
  }
  __syncthreads();

  // ---- Stage 1: QKV = xw * Wqkv^T + b  (M=64, N=768, K=256) ----
  {
    v16bf aF[8];
    #pragma unroll
    for (int kk = 0; kk < 8; ++kk) aF[kk] = loadA(xw, SX, mtF * 16, kk * 32, lane);

    for (int nt = ntP; nt < 48; nt += 2) {        // 24 N-tiles per wave, fixed A
      v8f acc = {};
      #pragma unroll
      for (int kk = 0; kk < 8; ++kk) {
        v16bf bb = loadB(wts + WQKV_OFF, Cc, nt * 16, kk * 32, lane);
        acc = __builtin_amdgcn_wmma_f32_16x16x32_bf16(false, aF[kk], false, bb, (short)0, acc, false, false);
      }
      float bias = bf2f(wts[BQKV_OFF + nt * 16 + (lane & 15)]);
      int seg  = nt >> 4;                         // 0=Q 1=K 2=V
      int cn   = ((nt & 15) << 4) + (lane & 15);  // channel within segment
      int mOff = (mtF << 4) + ((lane >> 4) << 3); // C/D layout: 8 consecutive M rows per lane
      if (seg == 2) {
        // V stored transposed: 8 consecutive tokens in row `cn` -> one packed b128 store
        uint16_t h[8];
        #pragma unroll
        for (int r = 0; r < 8; ++r) h[r] = f2bf(acc[r] + bias);
        uint4 pk;
        pk.x = (uint32_t)h[0] | ((uint32_t)h[1] << 16);
        pk.y = (uint32_t)h[2] | ((uint32_t)h[3] << 16);
        pk.z = (uint32_t)h[4] | ((uint32_t)h[5] << 16);
        pk.w = (uint32_t)h[6] | ((uint32_t)h[7] << 16);
        *(uint4*)(Vt + cn * SV + mOff) = pk;
      } else {
        uint16_t* dst = (seg == 0) ? Qm : Km;
        #pragma unroll
        for (int r = 0; r < 8; ++r) dst[(mOff + r) * SX + cn] = f2bf(acc[r] + bias);
      }
    }
  }
  __syncthreads();

  // ---- Stage 2: S = Q K^T / sqrt(256)  (M=64, N=64, K=256) ----
  {
    v16bf aF[8];
    #pragma unroll
    for (int kk = 0; kk < 8; ++kk) aF[kk] = loadA(Qm, SX, mtF * 16, kk * 32, lane);

    for (int kt = ntP; kt < 4; kt += 2) {         // 2 key tiles per wave
      v8f acc = {};
      #pragma unroll
      for (int kk = 0; kk < 8; ++kk) {
        v16bf bb = loadB(Km, SX, kt * 16, kk * 32, lane);  // B col = key token, contiguous channels
        acc = __builtin_amdgcn_wmma_f32_16x16x32_bf16(false, aF[kk], false, bb, (short)0, acc, false, false);
      }
      int n    = (kt << 4) + (lane & 15);
      int mOff = (mtF << 4) + ((lane >> 4) << 3);
      #pragma unroll
      for (int r = 0; r < 8; ++r) Sm[(mOff + r) * SS + n] = acc[r] * 0.0625f;
    }
  }
  __syncthreads();

  // ---- Stage 3: row softmax over 49 valid cols -> P bf16 (cols 49..63 = 0) ----
  if (tid < TP) {
    float* srow = Sm + tid * SS;
    uint16_t* prow = Pm + tid * SP;
    float mx = -3.4e38f;
    for (int j = 0; j < TOK; ++j) mx = fmaxf(mx, srow[j]);
    float sum = 0.0f;
    for (int j = 0; j < TOK; ++j) { float e = __expf(srow[j] - mx); srow[j] = e; sum += e; }
    float inv = 1.0f / sum;
    for (int j = 0; j < TOK; ++j) prow[j] = f2bf(srow[j] * inv);
    for (int j = TOK; j < TP; ++j) prow[j] = 0;
  }
  __syncthreads();

  // ---- Stage 4: O = P * V  (M=64, N=256, K=64) -> reuse xw buffer ----
  uint16_t* Om = xw;
  {
    v16bf aF[2];
    #pragma unroll
    for (int kk = 0; kk < 2; ++kk) aF[kk] = loadA(Pm, SP, mtF * 16, kk * 32, lane);

    for (int nt = ntP; nt < 16; nt += 2) {        // 8 channel tiles per wave
      v8f acc = {};
      #pragma unroll
      for (int kk = 0; kk < 2; ++kk) {
        v16bf bb = loadB(Vt, SV, nt * 16, kk * 32, lane);  // V^T row = channel, contiguous tokens
        acc = __builtin_amdgcn_wmma_f32_16x16x32_bf16(false, aF[kk], false, bb, (short)0, acc, false, false);
      }
      int cn   = (nt << 4) + (lane & 15);
      int mOff = (mtF << 4) + ((lane >> 4) << 3);
      #pragma unroll
      for (int r = 0; r < 8; ++r) Om[(mOff + r) * SX + cn] = f2bf(acc[r]);
    }
  }
  __syncthreads();

  // ---- Stage 5: out = O * Wproj^T + b, scatter fp32 to NCHW ----
  {
    v16bf aF[8];
    #pragma unroll
    for (int kk = 0; kk < 8; ++kk) aF[kk] = loadA(Om, SX, mtF * 16, kk * 32, lane);

    for (int nt = ntP; nt < 16; nt += 2) {        // 8 output-channel tiles per wave
      v8f acc = {};
      #pragma unroll
      for (int kk = 0; kk < 8; ++kk) {
        v16bf bb = loadB(wts + WPROJ_OFF, Cc, nt * 16, kk * 32, lane);
        acc = __builtin_amdgcn_wmma_f32_16x16x32_bf16(false, aF[kk], false, bb, (short)0, acc, false, false);
      }
      float bias = bf2f(wts[BPROJ_OFF + nt * 16 + (lane & 15)]);
      int cn   = (nt << 4) + (lane & 15);
      int mOff = (mtF << 4) + ((lane >> 4) << 3);
      #pragma unroll
      for (int r = 0; r < 8; ++r) {
        int m = mOff + r;
        if (m < TOK) {
          int hh = h0 + m / WSZ;
          int w2 = w0 + m % WSZ;
          out[((b * Cc + cn) * Hc + hh) * Wc + w2] = acc[r] + bias;
        }
      }
    }
  }
}

extern "C" void kernel_launch(void* const* d_in, const int* in_sizes, int n_in,
                              void* d_out, int out_size, void* d_ws, size_t ws_size,
                              hipStream_t stream) {
  const float* x     = (const float*)d_in[0];
  const float* wqkv  = (const float*)d_in[1];
  const float* bqkv  = (const float*)d_in[2];
  const float* wproj = (const float*)d_in[3];
  const float* bproj = (const float*)d_in[4];
  uint16_t* wts = (uint16_t*)d_ws;                // bf16 weight cache (526 KB)
  float* out = (float*)d_out;

  la_prep_weights<<<(WTS_ELEMS + 255) / 256, 256, 0, stream>>>(wqkv, bqkv, wproj, bproj, wts);

  const size_t ldsBytes = (size_t)(3 * TP * SX + Cc * SV) * 2 + (size_t)TP * SS * 4 + (size_t)TP * SP * 2; // 163840
  la_window_attn<<<4096, 256, ldsBytes, stream>>>(x, wts, out);
}
